// SelfAttnROI_1142461300795
// MI455X (gfx1250) — matrix operand order repaired and checked by reading, not compiled, not run on hardware
//
#include <hip/hip_runtime.h>

// ---------------- types / helpers ----------------
typedef unsigned short u16;
typedef __attribute__((ext_vector_type(16))) __bf16 v16bf;
typedef __attribute__((ext_vector_type(8)))  float  v8f;
typedef __attribute__((ext_vector_type(4)))  unsigned int v4u;
typedef __attribute__((ext_vector_type(8)))  int v8i;
typedef __attribute__((ext_vector_type(4)))  int v4i;

__device__ __forceinline__ u16 f32_to_bf16_rne(float f) {
  union { float f; unsigned u; } c; c.f = f;
  unsigned u = c.u;
  u += 0x7FFFu + ((u >> 16) & 1u);   // round-to-nearest-even
  return (u16)(u >> 16);
}

#define B_ROWS  2048
#define TOTAL_X 24696
#define R_N     333
#define MAXI    119
#define MAXO    12
#define CONCAT  3325
#define E_DIM   3328

// ---------------- ROI metadata (prefix sums) ----------------
__global__ void meta_kernel(int* __restrict__ xstart, int* __restrict__ outstart) {
  if (threadIdx.x == 0 && blockIdx.x == 0) {
    int xs = 0, os = 0;
    for (int r = 0; r < R_N; ++r) {
      xstart[r] = xs;
      outstart[r] = os;
      int sz = 30 + (7 * r) % 90;
      int rd = (r == 332) ? 8 : (8 + r % 5);
      xs += sz;
      os += rd;
    }
  }
}

// ---------------- ragged per-ROI linear encode ----------------
__global__ void roi_encode_kernel(const float* __restrict__ x,
                                  const float* __restrict__ encW,
                                  const float* __restrict__ encB,
                                  const int* __restrict__ xstart,
                                  const int* __restrict__ outstart,
                                  float* __restrict__ yraw) {
  extern __shared__ float smem[];
  float* xs  = smem;              // [64][MAXI]
  float* wsm = smem + 64 * MAXI;  // [MAXO][MAXI]
  const int rr  = blockIdx.x;
  const int b0  = blockIdx.y * 64;
  const int s   = 30 + (7 * rr) % 90;
  const int m   = (rr == 332) ? 8 : (8 + rr % 5);
  const int xst = xstart[rr];
  const int ost = outstart[rr];
  const int tid = threadIdx.x;

  for (int idx = tid; idx < 64 * s; idx += 256) {
    int row = idx / s, i = idx - row * s;
    xs[row * MAXI + i] = x[(size_t)(b0 + row) * TOTAL_X + xst + i];
  }
  for (int idx = tid; idx < m * s; idx += 256) {
    int o = idx / s, i = idx - o * s;
    wsm[o * MAXI + i] = encW[((size_t)rr * MAXO + o) * MAXI + i];
  }
  __syncthreads();
  for (int idx = tid; idx < 64 * m; idx += 256) {
    int row = idx & 63, o = idx >> 6;
    float acc = encB[rr * MAXO + o];
    const float* xr = &xs[row * MAXI];
    const float* wr = &wsm[o * MAXI];
    for (int i = 0; i < s; ++i) acc = fmaf(xr[i], wr[i], acc);
    yraw[(size_t)(b0 + row) * E_DIM + ost + o] = acc;
  }
}

// ---------------- batch-norm column statistics ----------------
__global__ void bn_stats_kernel(const float* __restrict__ src, int ld, int rows, int cols,
                                float* __restrict__ mean, float* __restrict__ rstd) {
  int c = blockIdx.x * blockDim.x + threadIdx.x;
  if (c >= cols) return;
  float s = 0.f, ss = 0.f;
  for (int r = 0; r < rows; ++r) {
    float v = src[(size_t)r * ld + c];
    s += v; ss += v * v;
  }
  float mu  = s / (float)rows;
  float var = ss / (float)rows - mu * mu;
  mean[c] = mu;
  rstd[c] = rsqrtf(var + 1e-5f);
}

// BN1 + LReLU(0.3) + F.pad(1,2)  ->  bf16 activation [B, E_DIM]
__global__ void norm_pad_kernel(const float* __restrict__ yraw,
                                const float* __restrict__ mean, const float* __restrict__ rstd,
                                const float* __restrict__ g, const float* __restrict__ bta,
                                u16* __restrict__ A1) {
  size_t idx = (size_t)blockIdx.x * blockDim.x + threadIdx.x;
  if (idx >= (size_t)B_ROWS * E_DIM) return;
  int c = (int)(idx % E_DIM);
  int b = (int)(idx / E_DIM);
  float out = 0.f;
  if (c >= 1 && c <= CONCAT) {
    int j = c - 1;
    float v = (yraw[(size_t)b * E_DIM + j] - mean[j]) * rstd[j] * g[j] + bta[j];
    out = v > 0.f ? v : 0.3f * v;
  }
  A1[idx] = f32_to_bf16_rne(out);
}

// BN2 + LReLU(slope) -> bf16
__global__ void norm_lrelu_kernel(const float* __restrict__ h,
                                  const float* __restrict__ mean, const float* __restrict__ rstd,
                                  const float* __restrict__ g, const float* __restrict__ bta,
                                  u16* __restrict__ out, int cols, float slope) {
  size_t idx = (size_t)blockIdx.x * blockDim.x + threadIdx.x;
  if (idx >= (size_t)B_ROWS * cols) return;
  int c = (int)(idx % cols);
  float v = (h[idx] - mean[c]) * rstd[c] * g[c] + bta[c];
  v = v > 0.f ? v : slope * v;
  out[idx] = f32_to_bf16_rne(v);
}

// ---------------- Tensor Data Mover: 2-D bf16 tile -> LDS (padded rows) ----------------
// tile: 32 elements (64B) x 128 rows, row stride = lda elements.
// LDS pad: after every 16 DWORDs (64B) insert 4 DWORDs (16B) -> 80B row stride == LDSB.
__device__ __forceinline__ void tdm_load_tile_a(const u16* gaddr, unsigned lds_off, int lda) {
  unsigned long long ga = (unsigned long long)gaddr;
  v4u g0;
  g0[0] = 1u;                                    // count=1, user descriptor
  g0[1] = lds_off;                               // lds_addr (bytes)
  g0[2] = (unsigned)(ga & 0xFFFFFFFFu);          // global_addr[31:0]
  g0[3] = (unsigned)((ga >> 32) & 0x1FFFFFFu)    // global_addr[56:32]
        | (2u << 30);                            // type=2 ("image")
  v8i g1;
  unsigned td0 = (unsigned)lda;
  g1[0] = (int)((1u << 16)      // data_size = 2 bytes
              | (1u << 20)      // pad_enable
              | (3u << 22)      // pad_interval: 16 DWORDs
              | (3u << 25));    // pad_amount: 4 DWORDs
  g1[1] = (int)((td0 & 0xFFFFu) << 16);          // tensor_dim0[15:0] @ bits 63:48
  g1[2] = (int)((td0 >> 16) | (2048u << 16));    // tensor_dim0 hi, tensor_dim1 lo (=2048)
  g1[3] = (int)(32u << 16);                      // tensor_dim1 hi=0, tile_dim0=32
  g1[4] = (int)(128u);                           // tile_dim1=128, tile_dim2=0
  g1[5] = (int)td0;                              // tensor_dim0_stride lo32
  g1[6] = 0;                                     // stride0 hi, stride1 lo
  g1[7] = 0;
  v4i z4 = {0, 0, 0, 0};
#if defined(__clang_major__) && (__clang_major__ >= 23)
  v8i z8 = {0, 0, 0, 0, 0, 0, 0, 0};
  __builtin_amdgcn_tensor_load_to_lds(g0, g1, z4, z4, z8, 0);
#else
  __builtin_amdgcn_tensor_load_to_lds(g0, g1, z4, z4, 0);
#endif
}

// ---------------- BF16 WMMA GEMM:  C[M,N] = A[M,K](bf16) * W[N,K](f32->bf16)^T + bias ----
#define TM 128
#define TN 128
#define TK 32
#define LDSB 40  // padded K-stride in LDS (u16); 80B keeps 16B-aligned chunks

enum { EP_BF16 = 0, EP_F32 = 1, EP_BOTH = 2 };

template <int EP, bool GB>
__global__ __launch_bounds__(256) void gemm_nt_wmma(
    const u16* __restrict__ A, int lda,
    const float* __restrict__ Bw, int ldb, int Nvalid, int Kvalid,
    const float* __restrict__ bias, int nbias,
    float* __restrict__ Cf, int ldcf, int ncf,
    u16* __restrict__ Cb, int ldcb,
    int K) {
  __shared__ u16 As[2][TM][LDSB];
  __shared__ u16 Bs[2][TN][LDSB];
  const int tid  = threadIdx.x;
  const int lane = tid & 31;
  const int wave = tid >> 5;
  const int wr   = wave & 3;   // 4 wave-rows * 32 = 128
  const int wc   = wave >> 2;  // 2 wave-cols * 64 = 128
  const int hl   = lane >> 4;
  const int l16  = lane & 15;
  const int m0   = blockIdx.y * TM;
  const int n0   = blockIdx.x * TN;

  v8f acc[2][4] = {};

  const int brow = tid >> 1;        // 0..127
  const int bseg = (tid & 1) * 16;  // 0 | 16 elements
  const int nk   = K / TK;

  // --- loaders ---
  auto issueA = [&](int kt, int buf) {   // TDM DMA: A tile -> LDS (wave 0 issues)
    if (wave == 0) {
      tdm_load_tile_a(A + (size_t)m0 * lda + kt * TK,
                      (unsigned)(unsigned long long)(void*)&As[buf][0][0], lda);
    }
  };
  auto loadB = [&](int kt, int buf) {    // f32 weights -> bf16 LDS (all 256 threads)
    const int kb = kt * TK;
    const int n  = n0 + brow;
    union { u16 h[16]; uint4 q[2]; } pk;
    if (!GB || (n < Nvalid && (kb + bseg + 16) <= Kvalid)) {
      const float* gb = Bw + (size_t)n * ldb + kb + bseg;
      const float4 f0 = *(const float4*)(gb + 0);
      const float4 f1 = *(const float4*)(gb + 4);
      const float4 f2 = *(const float4*)(gb + 8);
      const float4 f3 = *(const float4*)(gb + 12);
      pk.h[0]=f32_to_bf16_rne(f0.x);  pk.h[1]=f32_to_bf16_rne(f0.y);
      pk.h[2]=f32_to_bf16_rne(f0.z);  pk.h[3]=f32_to_bf16_rne(f0.w);
      pk.h[4]=f32_to_bf16_rne(f1.x);  pk.h[5]=f32_to_bf16_rne(f1.y);
      pk.h[6]=f32_to_bf16_rne(f1.z);  pk.h[7]=f32_to_bf16_rne(f1.w);
      pk.h[8]=f32_to_bf16_rne(f2.x);  pk.h[9]=f32_to_bf16_rne(f2.y);
      pk.h[10]=f32_to_bf16_rne(f2.z); pk.h[11]=f32_to_bf16_rne(f2.w);
      pk.h[12]=f32_to_bf16_rne(f3.x); pk.h[13]=f32_to_bf16_rne(f3.y);
      pk.h[14]=f32_to_bf16_rne(f3.z); pk.h[15]=f32_to_bf16_rne(f3.w);
    } else {
      for (int i = 0; i < 16; ++i) {
        int k = kb + bseg + i;
        float v = (n < Nvalid && k < Kvalid) ? Bw[(size_t)n * ldb + k] : 0.f;
        pk.h[i] = f32_to_bf16_rne(v);
      }
    }
    *(uint4*)&Bs[buf][brow][bseg]     = pk.q[0];
    *(uint4*)&Bs[buf][brow][bseg + 8] = pk.q[1];
  };

  // --- software pipeline: DMA/convert tile kt+1 while computing tile kt ---
  issueA(0, 0);
  loadB(0, 0);
  // TENSORcnt is per-wave: no-op for waves 1..7, fences wave 0's DMA. Unconditional
  // avoids an exec-mask save/restore in the hot loop.
  __builtin_amdgcn_s_wait_tensorcnt(0);
  __syncthreads();

  int buf = 0;
  for (int kt = 0; kt < nk; ++kt) {
    if (kt + 1 < nk) {
      issueA(kt + 1, buf ^ 1);
      loadB(kt + 1, buf ^ 1);
    }
    // ---- fragments (ISA 16-bit A 16x32 / B 32x16 layouts) + WMMA ----
    union Frag { uint4 u[2]; v16bf v; } fa[2], fb[4];
    for (int mi = 0; mi < 2; ++mi) {
      const u16* pa = &As[buf][wr * 32 + mi * 16 + l16][0];
      fa[mi].u[0] = *(const uint4*)(pa + 8 * hl);
      fa[mi].u[1] = *(const uint4*)(pa + 16 + 8 * hl);
    }
    for (int ni = 0; ni < 4; ++ni) {
      const u16* pb = &Bs[buf][wc * 64 + ni * 16 + l16][0];
      fb[ni].u[0] = *(const uint4*)(pb + 8 * hl);
      fb[ni].u[1] = *(const uint4*)(pb + 16 + 8 * hl);
    }
    for (int mi = 0; mi < 2; ++mi)
      for (int ni = 0; ni < 4; ++ni)
        acc[mi][ni] = __builtin_amdgcn_wmma_f32_16x16x32_bf16(
            false, fa[mi].v, false, fb[ni].v, (short)0, acc[mi][ni], false, false);
    __builtin_amdgcn_s_wait_tensorcnt(0);
    __syncthreads();
    buf ^= 1;
  }

  // ---- epilogue (C layout: N = lane%16, M = vgpr + 8*(lane/16)) ----
  for (int mi = 0; mi < 2; ++mi) {
    for (int ni = 0; ni < 4; ++ni) {
      const int coln = n0 + wc * 64 + ni * 16 + l16;
      const float bi = (coln < nbias) ? bias[coln] : 0.f;
      for (int gidx = 0; gidx < 8; ++gidx) {
        const int rowm = m0 + wr * 32 + mi * 16 + hl * 8 + gidx;
        const float v = acc[mi][ni][gidx] + bi;
        if (EP == EP_F32 || EP == EP_BOTH) {
          if (coln < ncf) Cf[(size_t)rowm * ldcf + coln] = v;
        }
        if (EP == EP_BF16 || EP == EP_BOTH) {
          Cb[(size_t)rowm * ldcb + coln] = f32_to_bf16_rne(v);
        }
      }
    }
  }
}

// ---------------- row softmax over 180 classes ----------------
__global__ void softmax_kernel(const float* __restrict__ logits, int ld,
                               float* __restrict__ out) {
  __shared__ float red[256];
  const int b = blockIdx.x;
  const int t = threadIdx.x;
  float v = (t < 180) ? logits[(size_t)b * ld + t] : -3.0e38f;
  red[t] = v; __syncthreads();
  for (int s = 128; s > 0; s >>= 1) {
    if (t < s) red[t] = fmaxf(red[t], red[t + s]);
    __syncthreads();
  }
  const float mx = red[0]; __syncthreads();
  float e = (t < 180) ? expf(v - mx) : 0.f;
  red[t] = e; __syncthreads();
  for (int s = 128; s > 0; s >>= 1) {
    if (t < s) red[t] += red[t + s];
    __syncthreads();
  }
  const float inv = 1.f / red[0];
  if (t < 180) out[(size_t)b * 180 + t] = e * inv;
}

// ---------------- launch ----------------
extern "C" void kernel_launch(void* const* d_in, const int* in_sizes, int n_in,
                              void* d_out, int out_size, void* d_ws, size_t ws_size,
                              hipStream_t stream) {
  (void)in_sizes; (void)n_in; (void)out_size; (void)ws_size;
  const float* x     = (const float*)d_in[0];
  const float* enc_W = (const float*)d_in[1];
  const float* enc_b = (const float*)d_in[2];
  const float* bn1_g = (const float*)d_in[3];
  const float* bn1_b = (const float*)d_in[4];
  const float* Wqkv  = (const float*)d_in[5];
  const float* bqkv  = (const float*)d_in[6];
  const float* Wo    = (const float*)d_in[7];
  const float* bo    = (const float*)d_in[8];
  const float* W1    = (const float*)d_in[9];
  const float* b1    = (const float*)d_in[10];
  const float* bn2_g = (const float*)d_in[11];
  const float* bn2_b = (const float*)d_in[12];
  const float* W2    = (const float*)d_in[13];
  const float* b2    = (const float*)d_in[14];
  const float* Wc    = (const float*)d_in[15];
  const float* bc    = (const float*)d_in[16];

  char* base = (char*)d_ws;
  size_t off = 0;
  auto alloc = [&](size_t bytes) { size_t o = off; off = (off + bytes + 255) & ~(size_t)255; return o; };
  int*   xstart = (int*)  (base + alloc(R_N * 4));
  int*   ostart = (int*)  (base + alloc(R_N * 4));
  float* yraw   = (float*)(base + alloc((size_t)B_ROWS * E_DIM * 4));
  float* mean1  = (float*)(base + alloc(E_DIM * 4));
  float* rstd1  = (float*)(base + alloc(E_DIM * 4));
  u16*   A1     = (u16*)  (base + alloc((size_t)B_ROWS * E_DIM * 2));
  u16*   Vb     = (u16*)  (base + alloc((size_t)B_ROWS * E_DIM * 2));
  u16*   Ob     = (u16*)  (base + alloc((size_t)B_ROWS * E_DIM * 2));
  float* H1     = (float*)(base + alloc((size_t)B_ROWS * 1024 * 4));
  float* mean2  = (float*)(base + alloc(1024 * 4));
  float* rstd2  = (float*)(base + alloc(1024 * 4));
  u16*   Rb     = (u16*)  (base + alloc((size_t)B_ROWS * 1024 * 2));
  u16*   regb   = (u16*)  (base + alloc((size_t)B_ROWS * 384 * 2));
  float* logits = (float*)(base + alloc((size_t)B_ROWS * 256 * 4));

  float* out_reg = (float*)d_out;                  // [2048,300]
  float* out_sm  = out_reg + (size_t)B_ROWS * 300; // [2048,180]

  // 1) ROI metadata
  meta_kernel<<<1, 32, 0, stream>>>(xstart, ostart);

  // 2) ragged ROI encode -> yraw [2048, 3325] (stride E_DIM)
  {
    dim3 grid(R_N, B_ROWS / 64);
    size_t shmem = (size_t)(64 * MAXI + MAXO * MAXI) * 4;
    roi_encode_kernel<<<grid, 256, shmem, stream>>>(x, enc_W, enc_b, xstart, ostart, yraw);
  }

  // 3) BN1 stats + normalize/LReLU(0.3)/pad -> A1 bf16 [2048, 3328]
  bn_stats_kernel<<<(CONCAT + 255) / 256, 256, 0, stream>>>(yraw, E_DIM, B_ROWS, CONCAT, mean1, rstd1);
  norm_pad_kernel<<<(int)(((size_t)B_ROWS * E_DIM + 255) / 256), 256, 0, stream>>>(
      yraw, mean1, rstd1, bn1_g, bn1_b, A1);

  // 4) L=1 attention: softmax(q.k) == [1] exactly, so attn out == V = A1 @ Wv^T + bv
  gemm_nt_wmma<EP_BF16, false><<<dim3(E_DIM / TN, B_ROWS / TM), 256, 0, stream>>>(
      A1, E_DIM, Wqkv + (size_t)2 * E_DIM * E_DIM, E_DIM, E_DIM, E_DIM,
      bqkv + 2 * E_DIM, E_DIM, nullptr, 0, 0, Vb, E_DIM, E_DIM);

  // 5) O = V @ Wo^T + bo
  gemm_nt_wmma<EP_BF16, false><<<dim3(E_DIM / TN, B_ROWS / TM), 256, 0, stream>>>(
      Vb, E_DIM, Wo, E_DIM, E_DIM, E_DIM, bo, E_DIM, nullptr, 0, 0, Ob, E_DIM, E_DIM);

  // 6) H1 = O @ W1^T + b1  (f32, for BN2 batch stats)
  gemm_nt_wmma<EP_F32, false><<<dim3(1024 / TN, B_ROWS / TM), 256, 0, stream>>>(
      Ob, E_DIM, W1, E_DIM, 1024, E_DIM, b1, 1024, H1, 1024, 1024, nullptr, 0, E_DIM);

  // 7) BN2 + LReLU(0.1) -> Rb bf16 [2048,1024]
  bn_stats_kernel<<<1024 / 256, 256, 0, stream>>>(H1, 1024, B_ROWS, 1024, mean2, rstd2);
  norm_lrelu_kernel<<<(int)(((size_t)B_ROWS * 1024 + 255) / 256), 256, 0, stream>>>(
      H1, mean2, rstd2, bn2_g, bn2_b, Rb, 1024, 0.1f);

  // 8) reg_out = Rb @ W2^T + b2 : f32 -> d_out[:,0:300], bf16 (N padded 384) -> regb
  gemm_nt_wmma<EP_BOTH, true><<<dim3(384 / TN, B_ROWS / TM), 256, 0, stream>>>(
      Rb, 1024, W2, 1024, 300, 1024, b2, 300, out_reg, 300, 300, regb, 384, 1024);

  // 9) logits = regb @ Wc^T + bc  (N padded 256, K padded 384; regb cols 300.. are 0)
  gemm_nt_wmma<EP_F32, true><<<dim3(256 / TN, B_ROWS / TM), 256, 0, stream>>>(
      regb, 384, Wc, 300, 180, 300, bc, 180, logits, 256, 256, nullptr, 0, 384);

  // 10) softmax -> d_out tail
  softmax_kernel<<<B_ROWS, 256, 0, stream>>>(logits, 256, out_sm);
}
